// ProxyMemoryBank_22574348107947
// MI455X (gfx1250) — matrix operand order repaired and checked by reading, not compiled, not run on hardware
//
#include <hip/hip_runtime.h>
#include <hip/hip_bf16.h>

// ---------------------------------------------------------------------------
// Problem constants (match reference)
// ---------------------------------------------------------------------------
#define BB   256          // batch
#define DD   2048         // feature dim
#define PP   8192         // proxies
#define CC   8            // cameras
#define TEMP 0.07f
#define MOM  0.2f
#define NEGINF (-1e30f)

typedef __attribute__((ext_vector_type(16))) __bf16 bf16x16;
typedef __attribute__((ext_vector_type(8)))  float  v8f;

union Frag {
    bf16x16 v;
    uint4   q[2];
};

// ---------------------------------------------------------------------------
// 1) fp32 -> bf16 (round-to-nearest-even)
// ---------------------------------------------------------------------------
__global__ void cvt_f32_bf16(const float* __restrict__ in,
                             unsigned short* __restrict__ out, int n) {
    int i = blockIdx.x * blockDim.x + threadIdx.x;
    if (i < n) {
        unsigned u = __float_as_uint(in[i]);
        out[i] = (unsigned short)((u + 0x7FFFu + ((u >> 16) & 1u)) >> 16);
    }
}

// ---------------------------------------------------------------------------
// 2) zero the per-cam accumulators: [0..7]=ce, [8..15]=inter, [16..23]=count
// ---------------------------------------------------------------------------
__global__ void zero_accum(float* accum) {
    if (threadIdx.x < 24) accum[threadIdx.x] = 0.0f;
}

// ---------------------------------------------------------------------------
// 3) sim = A(256x2048,bf16) * M(8192x2048,bf16)^T
//    Double-buffered async-LDS pipeline + v_wmma_f32_16x16x32_bf16.
//
//    Workgroup tile: M=32, N=256, K-step=32. 8 waves; wave w computes
//    rows 16*(w/4) .. +16, cols 64*(w%4) .. +64 (4 WMMA accumulators).
//
//    LDS layout per buffer (rows padded to 80B -> bank-conflict-free b128):
//      A tile: 32 rows  x 32 K : row m at byte m*80          (2560 B)
//      B tile: 256 cols x 32 K : col c at byte 2560 + c*80   (20480 B)
//    buffer stride 23040 B, double buffered = 46080 B LDS.
// ---------------------------------------------------------------------------
#define BUF_USH   11520   // 23040 bytes / 2
#define BOFF_USH  1280    // 2560 bytes / 2
#define NCHUNK    1152    // 1024 B-chunks + 128 A-chunks, 16B each

// The LDS destination address is derived from the pointer itself (ptrtoint):
// low 32 bits of a generic pointer to a __shared__ object are its as(3)
// offset, which is what the async instruction wants (HW adds LDS_BASE).
// The ptrtoint also ESCAPES the shared array, so the "memory" clobber makes
// the compiler treat it as written by the DMA (otherwise the ds_load
// fragment reads get folded to undef -- observed in round 2).
__device__ __forceinline__ void async_b128_to_lds(const unsigned short* lds_elem,
                                                  const unsigned short* gptr) {
    const unsigned off = (unsigned)(size_t)lds_elem;
    asm volatile("global_load_async_to_lds_b128 %0, %1, off"
                 :: "v"(off), "v"(gptr)
                 : "memory");
}

__global__ __launch_bounds__(256)
void gemm_bf16_wmma(const unsigned short* __restrict__ Abf,
                    const unsigned short* __restrict__ Mbf,
                    float* __restrict__ sim) {
    __shared__ __align__(16) unsigned short lds[2 * BUF_USH];

    const int tid  = threadIdx.x;
    const int lane = tid & 31;
    const int wave = tid >> 5;
    const int h    = lane >> 4;      // half-wave (0/1)
    const int l16  = lane & 15;

    const int mBlock = blockIdx.x * 32;
    const int nBlock = blockIdx.y * 256;

    const int mLoc  = (wave >> 2) * 16 + l16;       // A fragment row (local)
    const int cBase = (wave & 3) * 64;              // B local col base

    // ---- stage K-tile `ks` into buffer `buf` (async DMA to LDS) ----
    auto stage = [&](int ks, int buf) {
        const int k0 = ks * 32;
        const unsigned bufUsh = (unsigned)buf * BUF_USH;
        for (int i = tid; i < NCHUNK; i += 256) {
            const unsigned short* lp;
            const unsigned short* g;
            if (i < 1024) {                 // B tile: 256 cols x 4 chunks
                const int col = i >> 2, kc = i & 3;
                g  = Mbf + (size_t)(nBlock + col) * DD + k0 + kc * 8;
                lp = &lds[bufUsh + BOFF_USH + (unsigned)col * 40u + (unsigned)kc * 8u];
            } else {                        // A tile: 32 rows x 4 chunks
                const int j = i - 1024;
                const int row = j >> 2, kc = j & 3;
                g  = Abf + (size_t)(mBlock + row) * DD + k0 + kc * 8;
                lp = &lds[bufUsh + (unsigned)row * 40u + (unsigned)kc * 8u];
            }
            async_b128_to_lds(lp, g);
        }
    };

    v8f acc[4] = {};

    stage(0, 0);
    for (int ks = 0; ks < DD / 32; ++ks) {
        const int buf = ks & 1;
        // wait for my DMA; the (template-unused) LDS operand re-clobbers `lds`
        // so fragment loads can never be hoisted across this point.
        asm volatile("s_wait_asynccnt 0x0"
                     :: "s"((unsigned)(size_t)&lds[0]) : "memory");
        __syncthreads();                                    // everyone's landed
        if (ks + 1 < DD / 32) stage(ks + 1, buf ^ 1);       // overlap next DMA

        const unsigned ub = (unsigned)buf * BUF_USH;

        // A fragment: q[0] = K 8h..8h+7, q[1] = K 16+8h..23+8h
        Frag a;
        const unsigned short* pa = &lds[ub + (unsigned)mLoc * 40u + (unsigned)h * 8u];
        a.q[0] = *(const uint4*)(pa);
        a.q[1] = *(const uint4*)(pa + 16);

#pragma unroll
        for (int t = 0; t < 4; ++t) {
            // B fragment: col = cBase+16t+l16, q[0] = K 8h..8h+7, q[1] = K 16+8h..23+8h
            Frag b;
            const int c = cBase + 16 * t + l16;
            const unsigned short* pb =
                &lds[ub + BOFF_USH + (unsigned)c * 40u + (unsigned)h * 8u];
            b.q[0] = *(const uint4*)(pb);
            b.q[1] = *(const uint4*)(pb + 16);

            acc[t] = __builtin_amdgcn_wmma_f32_16x16x32_bf16(
                false, a.v, false, b.v, (short)0, acc[t], false, false);
        }
        __syncthreads();    // all reads of this buffer done before its reuse
    }

    // C/D layout: VGPR r holds M = r + 8*(lane/16), N = lane%16
#pragma unroll
    for (int t = 0; t < 4; ++t) {
#pragma unroll
        for (int r = 0; r < 8; ++r) {
            const int m = mBlock + (wave >> 2) * 16 + r + 8 * h;
            const int n = nBlock + cBase + 16 * t + l16;
            sim[(size_t)m * PP + n] = acc[t][r];
        }
    }
}

// ---------------------------------------------------------------------------
// block-wide reductions (256 threads)
// ---------------------------------------------------------------------------
__device__ __forceinline__ float blockMax256(float v, float* red) {
    const int tid = threadIdx.x;
    red[tid] = v; __syncthreads();
    for (int s = 128; s > 0; s >>= 1) {
        if (tid < s) red[tid] = fmaxf(red[tid], red[tid + s]);
        __syncthreads();
    }
    float r = red[0]; __syncthreads();
    return r;
}
__device__ __forceinline__ float blockSum256(float v, float* red) {
    const int tid = threadIdx.x;
    red[tid] = v; __syncthreads();
    for (int s = 128; s > 0; s >>= 1) {
        if (tid < s) red[tid] += red[tid + s];
        __syncthreads();
    }
    float r = red[0]; __syncthreads();
    return r;
}
__device__ __forceinline__ int blockSumI256(int v, int* red) {
    const int tid = threadIdx.x;
    red[tid] = v; __syncthreads();
    for (int s = 128; s > 0; s >>= 1) {
        if (tid < s) red[tid] += red[tid + s];
        __syncthreads();
    }
    int r = red[0]; __syncthreads();
    return r;
}

// ---------------------------------------------------------------------------
// 4) per-row losses. One block (256 threads) per batch row.
// ---------------------------------------------------------------------------
__global__ __launch_bounds__(256)
void row_loss(const float* __restrict__ sim,
              const int* __restrict__ label,
              const int* __restrict__ camid,
              const int* __restrict__ pclabel,
              const int* __restrict__ proxy_cam,
              const int* __restrict__ proxy_clu,
              const int* __restrict__ epoch,
              const int* __restrict__ inter_epoch,
              const int* __restrict__ kptr,
              float* __restrict__ accum) {
    __shared__ float srow[PP];            // 32 KB: this row's raw dot products
    __shared__ unsigned char msk[PP];     // 8 KB
    __shared__ float red[256];
    __shared__ int  ired[256];

    const int row = blockIdx.x;
    const int tid = threadIdx.x;
    const int cam = camid[row];
    const int lbl = label[row];
    const int clu = pclabel[row];
    const float invT = 1.0f / TEMP;

    for (int p = tid; p < PP; p += 256) {
        srow[p] = sim[(size_t)row * PP + p];
        const int pc  = proxy_cam[p];
        const int pcl = proxy_clu[p];
        unsigned char m = 0;
        if (pcl != clu) m |= 1;                       // negative
        if (pcl == clu && pc != cam) m |= 2;          // positive
        if (pc == cam) m |= 4;                        // intra (same cam)
        msk[p] = m;
    }
    __syncthreads();

    // ----- intra-camera CE over this row's cam proxies -----
    float mx = NEGINF;
    for (int p = tid; p < PP; p += 256)
        if (msk[p] & 4) mx = fmaxf(mx, srow[p]);
    mx = blockMax256(mx, red);

    float se = 0.0f;
    for (int p = tid; p < PP; p += 256)
        if (msk[p] & 4) se += __expf((srow[p] - mx) * invT);
    se = blockSum256(se, red);

    const float lse_intra = mx * invT + __logf(se);
    if (tid == 0) {
        const float ce = lse_intra - srow[lbl] * invT;
        atomicAdd(&accum[cam], ce);
        atomicAdd(&accum[16 + cam], 1.0f);
    }

    // ----- inter-camera loss (gated on device-side epoch) -----
    if (*epoch >= *inter_epoch) {
        const int K = *kptr;

        // k-th largest negative similarity by bisection on count(>= mid)
        float lo = -4.0f, hi = 4.0f;
        for (int it = 0; it < 40; ++it) {
            const float mid = 0.5f * (lo + hi);
            int c = 0;
            for (int p = tid; p < PP; p += 256)
                if ((msk[p] & 1) && srow[p] >= mid) ++c;
            c = blockSumI256(c, ired);
            if (c >= K) lo = mid; else hi = mid;
        }
        const float thresh = lo;

        // full mask = positives | (negatives strictly below the k-th largest)
        float mx2 = NEGINF;
        for (int p = tid; p < PP; p += 256) {
            const bool keep = (msk[p] & 2) || ((msk[p] & 1) && srow[p] < thresh);
            if (keep) mx2 = fmaxf(mx2, srow[p]);
        }
        mx2 = blockMax256(mx2, red);

        float se2 = 0.0f;
        for (int p = tid; p < PP; p += 256) {
            const bool keep = (msk[p] & 2) || ((msk[p] & 1) && srow[p] < thresh);
            if (keep) se2 += __expf((srow[p] - mx2) * invT);
        }
        se2 = blockSum256(se2, red);
        const float lse = mx2 * invT + __logf(se2);

        float psum = 0.0f; int pcard = 0;
        for (int p = tid; p < PP; p += 256)
            if (msk[p] & 2) { psum += srow[p] * invT - lse; ++pcard; }
        psum  = blockSum256(psum, red);
        pcard = blockSumI256(pcard, ired);

        if (tid == 0 && pcard > 0)
            atomicAdd(&accum[8 + cam], -psum / (float)pcard);
    }
}

// ---------------------------------------------------------------------------
// 5) finalize scalar loss
// ---------------------------------------------------------------------------
__global__ void finalize_loss(const float* __restrict__ accum,
                              float* __restrict__ out) {
    if (threadIdx.x == 0) {
        float loss = 0.0f;
        for (int c = 0; c < CC; ++c) {
            const float cnt = accum[16 + c];
            if (cnt > 0.0f)
                loss += accum[c] / cnt + 0.5f * accum[8 + c] / cnt;
        }
        out[0] = loss;
    }
}

// ---------------------------------------------------------------------------
// 6) chain-parallel EMA memory-bank update.
//    Block b handles batch index b ONLY if it is the first occurrence of its
//    label; it then walks the whole duplicate-label chain in-order, keeping
//    the row in registers (one LDS norm-reduction per chain step).
// ---------------------------------------------------------------------------
__global__ __launch_bounds__(256)
void ema_update_chains(const float* __restrict__ feat,
                       const float* __restrict__ memory0,
                       const int* __restrict__ label,
                       float* __restrict__ memOut) {
    __shared__ float red[256];
    const int i   = blockIdx.x;
    const int tid = threadIdx.x;
    const int y   = label[i];

    // block-uniform predicate: am I the first occurrence of label y?
    for (int j = 0; j < i; ++j)
        if (label[j] == y) return;       // uniform early-out, no barrier issues

    float v[8];
    const float* src = memory0 + (size_t)y * DD;
#pragma unroll
    for (int jj = 0; jj < 8; ++jj) v[jj] = src[tid * 8 + jj];

    for (int j = i; j < BB; ++j) {
        if (label[j] != y) continue;     // uniform
        const float* x = feat + (size_t)j * DD;
        float ss = 0.0f;
#pragma unroll
        for (int jj = 0; jj < 8; ++jj) {
            v[jj] = MOM * v[jj] + (1.0f - MOM) * x[tid * 8 + jj];
            ss += v[jj] * v[jj];
        }
        red[tid] = ss; __syncthreads();
        for (int s = 128; s > 0; s >>= 1) {
            if (tid < s) red[tid] += red[tid + s];
            __syncthreads();
        }
        const float inv = rsqrtf(fmaxf(red[0], 1e-24f));
        __syncthreads();
#pragma unroll
        for (int jj = 0; jj < 8; ++jj) v[jj] *= inv;
    }

    float* dst = memOut + (size_t)y * DD;
#pragma unroll
    for (int jj = 0; jj < 8; ++jj) dst[tid * 8 + jj] = v[jj];
}

// ---------------------------------------------------------------------------
// launch
// ---------------------------------------------------------------------------
extern "C" void kernel_launch(void* const* d_in, const int* in_sizes, int n_in,
                              void* d_out, int out_size, void* d_ws, size_t ws_size,
                              hipStream_t stream) {
    const float* feat      = (const float*)d_in[0];
    const float* memory    = (const float*)d_in[1];
    const int*   label     = (const int*)d_in[2];
    const int*   camid     = (const int*)d_in[3];
    const int*   pclabel   = (const int*)d_in[4];
    const int*   proxy_cam = (const int*)d_in[5];
    const int*   proxy_clu = (const int*)d_in[6];
    const int*   epoch     = (const int*)d_in[7];
    const int*   iepoch    = (const int*)d_in[8];
    const int*   kptr      = (const int*)d_in[9];
    float* out = (float*)d_out;

    char* ws = (char*)d_ws;
    unsigned short* Abf = (unsigned short*)ws;                         // 1 MB
    unsigned short* Mbf = (unsigned short*)(ws + (size_t)BB * DD * 2); // 32 MB
    float* sim   = (float*)(ws + (size_t)BB * DD * 2 + (size_t)PP * DD * 2); // 8 MB
    float* accum = (float*)((char*)sim + (size_t)BB * PP * sizeof(float));   // 96 B

    // 1) convert to bf16
    {
        const int nA = BB * DD;
        cvt_f32_bf16<<<(nA + 255) / 256, 256, 0, stream>>>(feat, Abf, nA);
        const int nM = PP * DD;
        cvt_f32_bf16<<<(nM + 255) / 256, 256, 0, stream>>>(memory, Mbf, nM);
    }
    // 2) zero accumulators
    zero_accum<<<1, 32, 0, stream>>>(accum);

    // 3) WMMA GEMM (async-LDS double-buffered)
    gemm_bf16_wmma<<<dim3(BB / 32, PP / 256), 256, 0, stream>>>(Abf, Mbf, sim);

    // 4) per-row losses
    row_loss<<<BB, 256, 0, stream>>>(sim, label, camid, pclabel,
                                     proxy_cam, proxy_clu,
                                     epoch, iepoch, kptr, accum);

    // 5) scalar loss
    finalize_loss<<<1, 32, 0, stream>>>(accum, out);

    // 6) new_memory = copy of bank, then chain-parallel EMA on touched rows
    hipMemcpyAsync(out + 1, memory, (size_t)PP * DD * sizeof(float),
                   hipMemcpyDeviceToDevice, stream);
    ema_update_chains<<<BB, 256, 0, stream>>>(feat, memory, label, out + 1);
}
